// skinnet_37306085933364
// MI455X (gfx1250) — matrix operand order repaired and checked by reading, not compiled, not run on hardware
//
#include <hip/hip_runtime.h>
#include <hip/hip_bf16.h>
#include <math.h>

// ---------------------------------------------------------------------------
// Types for CDNA5 WMMA (wave32, 16x16x32 bf16 -> f32)
// ---------------------------------------------------------------------------
typedef __attribute__((ext_vector_type(16))) __bf16 v16bf;
typedef __attribute__((ext_vector_type(8)))  __bf16 v8bf;
typedef __attribute__((ext_vector_type(8)))  float  v8f;

__device__ inline unsigned short bfbits(float f) {
    union { float f; unsigned u; } x; x.f = f;
    return (unsigned short)((x.u + 0x7FFFu + ((x.u >> 16) & 1u)) >> 16);  // RNE
}
__device__ inline __bf16 f2bf(float f) {
    union { unsigned short s; __bf16 b; } y; y.s = bfbits(f);
    return y.b;
}

// ---------------------------------------------------------------------------
// GEMM: C[M,N] = (relu?)(A[M,K] @ W[K,N] + bias)
// A: bf16 row-major (M x K).  Wt: bf16 N-major (Nn x K) = W transposed.
// Contract: K % 32 == 0, Nn % 64 == 0 (callers pad).  M arbitrary: A-row
// loads clamp branchlessly, stores guarded after all WMMAs (EXEC all-1s at
// every v_wmma).  Wave tile 32x32 (4 wmma / k-step); block = 8 waves as
// 4(M) x 2(N) => 128 x 64 block tile.  All fragment loads are contiguous
// vector loads; zero per-element conversion in the loop.
// ---------------------------------------------------------------------------
__global__ __launch_bounds__(256) void gemm_bf16_kernel(
    const __bf16* __restrict__ A, const __bf16* __restrict__ Wt,
    const float* __restrict__ bias, float* __restrict__ C,
    int M, int K, int Nn, int relu)
{
    const int wave = threadIdx.x >> 5;       // 0..7
    const int lane = threadIdx.x & 31;
    const int half = lane >> 4;
    const int lr   = lane & 15;
    const int wm   = wave >> 1;              // 0..3  (M direction)
    const int wn   = wave & 1;               // 0..1  (N direction)
    const int m0 = blockIdx.y * 128 + wm * 32;
    const int n0 = blockIdx.x * 64 + wn * 32;

    int mA0 = m0 + lr;       mA0 = (mA0 < M) ? mA0 : (M - 1);
    int mA1 = m0 + 16 + lr;  mA1 = (mA1 < M) ? mA1 : (M - 1);
    const __bf16* arow0 = A + (size_t)mA0 * K;
    const __bf16* arow1 = A + (size_t)mA1 * K;
    const int nB0 = n0 + lr;
    const int nB1 = n0 + 16 + lr;
    const __bf16* brow0 = Wt + (size_t)nB0 * K;
    const __bf16* brow1 = Wt + (size_t)nB1 * K;

    v8f acc00 = {}, acc01 = {}, acc10 = {}, acc11 = {};

    for (int k0 = 0; k0 < K; k0 += 32) {
        // A fragment: elements 0..7 at K=k0+half*8 (contiguous), 8..15 at +16.
        v8bf a0l = *(const v8bf*)(arow0 + k0 + half * 8);
        v8bf a0h = *(const v8bf*)(arow0 + k0 + 16 + half * 8);
        v8bf a1l = *(const v8bf*)(arow1 + k0 + half * 8);
        v8bf a1h = *(const v8bf*)(arow1 + k0 + 16 + half * 8);
        // B fragment: same K order as A (0..7 at k0+half*8, 8..15 at +16).
        v8bf b0l = *(const v8bf*)(brow0 + k0 + half * 8);
        v8bf b0h = *(const v8bf*)(brow0 + k0 + 16 + half * 8);
        v8bf b1l = *(const v8bf*)(brow1 + k0 + half * 8);
        v8bf b1h = *(const v8bf*)(brow1 + k0 + 16 + half * 8);
        v16bf b0 = __builtin_shufflevector(b0l, b0h, 0,1,2,3,4,5,6,7,8,9,10,11,12,13,14,15);
        v16bf b1 = __builtin_shufflevector(b1l, b1h, 0,1,2,3,4,5,6,7,8,9,10,11,12,13,14,15);
        v16bf a0 = __builtin_shufflevector(a0l, a0h, 0,1,2,3,4,5,6,7,8,9,10,11,12,13,14,15);
        v16bf a1 = __builtin_shufflevector(a1l, a1h, 0,1,2,3,4,5,6,7,8,9,10,11,12,13,14,15);
        acc00 = __builtin_amdgcn_wmma_f32_16x16x32_bf16(false, a0, false, b0, (short)0, acc00, false, false);
        acc01 = __builtin_amdgcn_wmma_f32_16x16x32_bf16(false, a0, false, b1, (short)0, acc01, false, false);
        acc10 = __builtin_amdgcn_wmma_f32_16x16x32_bf16(false, a1, false, b0, (short)0, acc10, false, false);
        acc11 = __builtin_amdgcn_wmma_f32_16x16x32_bf16(false, a1, false, b1, (short)0, acc11, false, false);
    }

    float bv0 = bias ? bias[nB0] : 0.f;
    float bv1 = bias ? bias[nB1] : 0.f;
#pragma unroll
    for (int r = 0; r < 8; ++r) {
        int row0 = m0 + half * 8 + r;        // C/D layout: VGPR r -> M=r (+8 high half)
        int row1 = row0 + 16;
        if (row0 < M) {
            float v0 = acc00[r] + bv0, v1 = acc01[r] + bv1;
            if (relu) { v0 = v0 > 0.f ? v0 : 0.f; v1 = v1 > 0.f ? v1 : 0.f; }
            C[(size_t)row0 * Nn + nB0] = v0;
            C[(size_t)row0 * Nn + nB1] = v1;
        }
        if (row1 < M) {
            float v0 = acc10[r] + bv0, v1 = acc11[r] + bv1;
            if (relu) { v0 = v0 > 0.f ? v0 : 0.f; v1 = v1 > 0.f ? v1 : 0.f; }
            C[(size_t)row1 * Nn + nB0] = v0;
            C[(size_t)row1 * Nn + nB1] = v1;
        }
    }
}

// ---------------------------------------------------------------------------
// Precision / layout transform kernels
// ---------------------------------------------------------------------------
// fp32 -> bf16, 2 elements per thread, packed 32-bit store
__global__ __launch_bounds__(256) void cvt_bf16_kernel(
    const float* __restrict__ in, __bf16* __restrict__ out, long long n2)
{
    long long i = (long long)blockIdx.x * 256 + threadIdx.x;
    if (i >= n2) return;
    float2 v = ((const float2*)in)[i];
    unsigned pk = (unsigned)bfbits(v.x) | ((unsigned)bfbits(v.y) << 16);
    ((unsigned*)out)[i] = pk;
}

// W (K x Nn fp32, row-major) -> Wt (Nn x K bf16)
__global__ __launch_bounds__(256) void transpose_w_bf16_kernel(
    const float* __restrict__ W, __bf16* __restrict__ Wt,
    long long total, int K, int Nn)
{
    long long i = (long long)blockIdx.x * 256 + threadIdx.x;
    if (i >= total) return;
    int n = (int)(i / K), k = (int)(i % K);
    Wt[i] = f2bf(W[(size_t)k * Nn + n]);
}

// ---------------------------------------------------------------------------
// Elementwise / reduction helper kernels
// ---------------------------------------------------------------------------
__global__ __launch_bounds__(256) void make_raw_kernel(
    const float* __restrict__ pos, const float* __restrict__ skin,
    float* __restrict__ raw, long long total, int SK)
{
    long long i = (long long)blockIdx.x * 256 + threadIdx.x;
    if (i >= total) return;
    int r = (int)(i / 64), c = (int)(i % 64);
    float v = 0.f;
    if (c < 3) v = pos[(size_t)r * 3 + c];
    else if (c < 3 + SK) v = skin[(size_t)r * SK + (c - 3)];
    raw[i] = v;
}

__global__ __launch_bounds__(256) void pad_rows_kernel(
    const float* __restrict__ in, float* __restrict__ out,
    long long total, int Kin, int Nn)
{
    long long i = (long long)blockIdx.x * 256 + threadIdx.x;
    if (i >= total) return;
    int r = (int)(i / Nn);
    out[i] = (r < Kin) ? in[i] : 0.f;
}

__global__ __launch_bounds__(256) void pad_cols_kernel(
    const float* __restrict__ in, float* __restrict__ out,
    long long total, int Cin, int Cp)
{
    long long i = (long long)blockIdx.x * 256 + threadIdx.x;
    if (i >= total) return;
    int r = (int)(i / Cp), c = (int)(i % Cp);
    out[i] = (c < Cin) ? in[(size_t)r * Cin + c] : 0.f;
}

__global__ __launch_bounds__(256) void extract_cols_kernel(
    const float* __restrict__ in, float* __restrict__ out,
    long long total, int Cout, int Cp)
{
    long long i = (long long)blockIdx.x * 256 + threadIdx.x;
    if (i >= total) return;
    int r = (int)(i / Cout), c = (int)(i % Cout);
    out[i] = in[(size_t)r * Cp + c];
}

__global__ __launch_bounds__(256) void sub_kernel(
    const float* __restrict__ A, const float* __restrict__ B,
    float* __restrict__ O, long long n)
{
    long long i = (long long)blockIdx.x * 256 + threadIdx.x;
    if (i < n) O[i] = A[i] - B[i];
}

__global__ __launch_bounds__(256) void fill_kernel(float* __restrict__ X, long long n, float v)
{
    long long i = (long long)blockIdx.x * 256 + threadIdx.x;
    if (i < n) X[i] = v;
}

__global__ __launch_bounds__(256) void fix_neginf_kernel(float* __restrict__ X, long long n)
{
    long long i = (long long)blockIdx.x * 256 + threadIdx.x;
    if (i < n && X[i] == -INFINITY) X[i] = 0.f;
}

__global__ __launch_bounds__(256) void concat2_kernel(
    const float* __restrict__ A, const float* __restrict__ B,
    float* __restrict__ O, long long total, int C1, int C2)
{
    long long i = (long long)blockIdx.x * 256 + threadIdx.x;
    if (i >= total) return;
    int C = C1 + C2;
    int r = (int)(i / C), c = (int)(i % C);
    O[i] = (c < C1) ? A[(size_t)r * C1 + c] : B[(size_t)r * C2 + (c - C1)];
}

__global__ __launch_bounds__(256) void gather_rows_kernel(
    const float* __restrict__ S, const int* __restrict__ idx,
    float* __restrict__ O, long long total, int C)
{
    long long i = (long long)blockIdx.x * 256 + threadIdx.x;
    if (i >= total) return;
    int r = (int)(i / C), c = (int)(i % C);
    O[i] = S[(size_t)idx[r] * C + c];
}

// m = relu(A1[dst] + B1[src] + bias)   (EdgeConv layer-1 after node precompute)
__global__ __launch_bounds__(256) void edge_combine_relu_kernel(
    const float* __restrict__ A1, const float* __restrict__ B1,
    const int* __restrict__ src, const int* __restrict__ dst,
    const float* __restrict__ bias, float* __restrict__ Mout,
    long long total, int H)
{
    long long i = (long long)blockIdx.x * 256 + threadIdx.x;
    if (i >= total) return;
    int e = (int)(i / H), h = (int)(i % H);
    float v = A1[(size_t)dst[e] * H + h] + B1[(size_t)src[e] * H + h] + bias[h];
    Mout[i] = v > 0.f ? v : 0.f;
}

// Deterministic two-stage column stats (mean / rsqrt(var+eps))
__global__ __launch_bounds__(256) void col_partial_kernel(
    const float* __restrict__ X, int M, int Nn,
    float* __restrict__ ps, float* __restrict__ pq)
{
    int c = blockIdx.y * 256 + threadIdx.x;
    if (c >= Nn) return;
    int r0 = blockIdx.x * 1024;
    int r1 = r0 + 1024; if (r1 > M) r1 = M;
    float s = 0.f, q = 0.f;
    for (int r = r0; r < r1; ++r) {
        float v = X[(size_t)r * Nn + c];
        s += v; q += v * v;
    }
    ps[(size_t)blockIdx.x * Nn + c] = s;
    pq[(size_t)blockIdx.x * Nn + c] = q;
}

__global__ __launch_bounds__(256) void col_finalize_kernel(
    const float* __restrict__ ps, const float* __restrict__ pq,
    int rb, int Nn, int M, float* __restrict__ mean, float* __restrict__ rstd)
{
    int c = blockIdx.x * 256 + threadIdx.x;
    if (c >= Nn) return;
    float s = 0.f, q = 0.f;
    for (int b = 0; b < rb; ++b) {
        s += ps[(size_t)b * Nn + c];
        q += pq[(size_t)b * Nn + c];
    }
    float mu  = s / (float)M;
    float var = q / (float)M - mu * mu;
    if (var < 0.f) var = 0.f;
    mean[c] = mu;
    rstd[c] = rsqrtf(var + 1e-5f);
}

__global__ __launch_bounds__(256) void bn_apply_kernel(
    float* __restrict__ X, const float* __restrict__ mean,
    const float* __restrict__ rstd, const float* __restrict__ g,
    const float* __restrict__ be, long long total, int Nn)
{
    long long i = (long long)blockIdx.x * 256 + threadIdx.x;
    if (i >= total) return;
    int c = (int)(i % Nn);
    X[i] = (X[i] - mean[c]) * rstd[c] * g[c] + be[c];
}

// order-independent (deterministic) float atomic max via int/uint atomics
__device__ inline void atomicMaxFloat(float* addr, float val)
{
    if (!(val == val)) return;
    if (val >= 0.f) atomicMax((int*)addr, __float_as_int(val));
    else            atomicMin((unsigned int*)addr, __float_as_uint(val));
}

__global__ __launch_bounds__(256) void seg_max_kernel(
    const float* __restrict__ X, const int* __restrict__ seg,
    float* __restrict__ O, long long total, int H)
{
    long long i = (long long)blockIdx.x * 256 + threadIdx.x;
    if (i >= total) return;
    int e = (int)(i / H), h = (int)(i % H);
    atomicMaxFloat(&O[(size_t)seg[e] * H + h], X[i]);
}

// ---------------------------------------------------------------------------
// Host-side orchestration
// ---------------------------------------------------------------------------
static inline int cdiv(long long a, long long b) { return (int)((a + b - 1) / b); }

struct Arena { char* p; size_t off; };
static void* araw(Arena& a, size_t bytes) {
    a.off = (a.off + 255) & ~(size_t)255;
    void* r = (void*)(a.p + a.off);
    a.off += bytes;
    return r;
}
static float*  aallocf(Arena& a, size_t n) { return (float*)araw(a, n * 4); }
static __bf16* aallocb(Arena& a, size_t n) { return (__bf16*)araw(a, n * 2); }

static void tobf(hipStream_t s, const float* in, __bf16* out, long long n) {
    long long n2 = n / 2;
    cvt_bf16_kernel<<<cdiv(n2, 256), 256, 0, s>>>(in, out, n2);
}

// W fp32 (K x Nn) is transposed+converted into wtb, then bf16 GEMM runs.
static void gemm(hipStream_t s, const __bf16* A, const float* W, const float* bias,
                 float* C, int M, int K, int Nn, int relu, __bf16* wtb)
{
    long long t = (long long)K * Nn;
    transpose_w_bf16_kernel<<<cdiv(t, 256), 256, 0, s>>>(W, wtb, t, K, Nn);
    dim3 g(cdiv(Nn, 64), cdiv(M, 128));
    gemm_bf16_kernel<<<g, 256, 0, s>>>(A, wtb, bias, C, M, K, Nn, relu);
}

static void bn(hipStream_t s, float* X, int M, int Nn,
               const float* g_, const float* be,
               float* mean, float* rstd, float* part)
{
    int rb = cdiv(M, 1024);
    dim3 gr(rb, cdiv(Nn, 256));
    col_partial_kernel<<<gr, 256, 0, s>>>(X, M, Nn, part, part + (size_t)rb * Nn);
    col_finalize_kernel<<<cdiv(Nn, 256), 256, 0, s>>>(part, part + (size_t)rb * Nn,
                                                      rb, Nn, M, mean, rstd);
    long long tot = (long long)M * Nn;
    bn_apply_kernel<<<cdiv(tot, 256), 256, 0, s>>>(X, mean, rstd, g_, be, tot, Nn);
}

static void linear_bn(hipStream_t s, const __bf16* xb, const float* W, const float* b,
                      const float* g_, const float* be, float* out,
                      int M, int K, int Nn, float* mean, float* rstd, float* part,
                      __bf16* wtb)
{
    gemm(s, xb, W, b, out, M, K, Nn, 1, wtb);
    bn(s, out, M, Nn, g_, be, mean, rstd, part);
}

// EdgeConv: m = MLP([x_i, x_j-x_i]); segment_max over dst; -inf -> 0.
// Layer 1 via node precompute: m1 = relu(A1[dst] + B1[src] + b0) where
// A1 = x @ (Wtop - Wbot), B1 = x @ Wbot  (8x FLOP reduction vs edge GEMM).
static void edge_conv(hipStream_t s, const __bf16* xb, int Nnodes, int C, int H,
                      const int* src, const int* dst, int E_,
                      const float* W0, const float* b0, const float* g0, const float* be0,
                      const float* W1, const float* b1, const float* g1, const float* be1,
                      float* out, float* A1, float* B1, float* e1, __bf16* e1b, float* e2,
                      float* wdiff, float* mean, float* rstd, float* part, __bf16* wtb)
{
    long long ch = (long long)C * H;
    sub_kernel<<<cdiv(ch, 256), 256, 0, s>>>(W0, W0 + ch, wdiff, ch);
    gemm(s, xb, wdiff,   nullptr, A1, Nnodes, C, H, 0, wtb);
    gemm(s, xb, W0 + ch, nullptr, B1, Nnodes, C, H, 0, wtb);
    long long eh = (long long)E_ * H;
    edge_combine_relu_kernel<<<cdiv(eh, 256), 256, 0, s>>>(A1, B1, src, dst, b0, e1, eh, H);
    bn(s, e1, E_, H, g0, be0, mean, rstd, part);
    tobf(s, e1, e1b, eh);
    gemm(s, e1b, W1, b1, e2, E_, H, H, 1, wtb);
    bn(s, e2, E_, H, g1, be1, mean, rstd, part);
    long long nh = (long long)Nnodes * H;
    fill_kernel<<<cdiv(nh, 256), 256, 0, s>>>(out, nh, -INFINITY);
    seg_max_kernel<<<cdiv(eh, 256), 256, 0, s>>>(e2, dst, out, eh, H);
    fix_neginf_kernel<<<cdiv(nh, 256), 256, 0, s>>>(out, nh);
}

extern "C" void kernel_launch(void* const* d_in, const int* in_sizes, int n_in,
                              void* d_out, int out_size, void* d_ws, size_t ws_size,
                              hipStream_t stream)
{
    (void)n_in; (void)out_size; (void)ws_size;
    const int N  = in_sizes[0] / 3;
    const int SK = in_sizes[1] / N;           // 40
    const float* pos  = (const float*)d_in[0];
    const float* skin = (const float*)d_in[1];
    auto P = [&](int i) { return (const float*)d_in[i]; };
    // pytree order: cls(2..15), gcu1(16..35), gcu2(36..55), gcu3(56..75),
    //               mlp1(76..83), mlp2(84..91)
    const int* tpl   = (const int*)d_in[92];
    const int* geo   = (const int*)d_in[93];
    const int* batch = (const int*)d_in[94];
    const int  E_ = in_sizes[92] / 2;
    const int  B_ = 8;
    const int* tpl_src = tpl;      const int* tpl_dst = tpl + E_;
    const int* geo_src = geo;      const int* geo_dst = geo + E_;

    Arena a{(char*)d_ws, 0};
    // fp32 buffers
    float* raw   = aallocf(a, (size_t)N * 64);
    float* w0p   = aallocf(a, (size_t)64 * 128);
    float* t128  = aallocf(a, (size_t)N * 128);
    float* x0    = aallocf(a, (size_t)N * 64);
    float* x1    = aallocf(a, (size_t)N * 256);
    float* x2    = aallocf(a, (size_t)N * 512);
    float* x3    = aallocf(a, (size_t)N * 512);
    float* A1    = aallocf(a, (size_t)N * 256);
    float* B1    = aallocf(a, (size_t)N * 256);
    float* convT = aallocf(a, (size_t)N * 256);
    float* convG = aallocf(a, (size_t)N * 256);
    float* catH  = aallocf(a, (size_t)N * 512);
    float* g     = aallocf(a, (size_t)N * 1024);
    float* gb    = aallocf(a, (size_t)N * 1024);
    float* gmax  = aallocf(a, (size_t)B_ * 1024);
    float* feat  = aallocf(a, (size_t)N * 1536);
    float* c1    = aallocf(a, (size_t)N * 1024);
    float* c2    = aallocf(a, (size_t)N * 512);
    float* c3    = aallocf(a, (size_t)N * 256);
    float* cout  = aallocf(a, (size_t)N * 64);
    float* wfp   = aallocf(a, (size_t)256 * 64);
    float* bfp   = aallocf(a, (size_t)64);
    float* e1    = aallocf(a, (size_t)E_ * 256);
    float* e2    = aallocf(a, (size_t)E_ * 256);
    float* wdiff = aallocf(a, (size_t)512 * 256);
    float* mean  = aallocf(a, 1024);
    float* rstd  = aallocf(a, 1024);
    float* part  = aallocf(a, 131072);
    // bf16 companions (GEMM inputs)
    __bf16* rawb  = aallocb(a, (size_t)N * 64);
    __bf16* t128b = aallocb(a, (size_t)N * 128);
    __bf16* x0b   = aallocb(a, (size_t)N * 64);
    __bf16* x1b   = aallocb(a, (size_t)N * 256);
    __bf16* x2b   = aallocb(a, (size_t)N * 512);
    __bf16* catHb = aallocb(a, (size_t)N * 512);
    __bf16* tmpb  = aallocb(a, (size_t)N * 256);
    __bf16* featb = aallocb(a, (size_t)N * 1536);
    __bf16* c1b   = aallocb(a, (size_t)N * 1024);
    __bf16* c2b   = aallocb(a, (size_t)N * 512);
    __bf16* c3b   = aallocb(a, (size_t)N * 256);
    __bf16* e1b   = aallocb(a, (size_t)E_ * 256);
    __bf16* wtb   = aallocb(a, (size_t)1536 * 1024);  // transposed-weight scratch

    // ---- mlp1: raw(43 pad 64) -> 128 -> 64 ----
    long long rt = (long long)N * 64;
    make_raw_kernel<<<cdiv(rt, 256), 256, 0, stream>>>(pos, skin, raw, rt, SK);
    tobf(stream, raw, rawb, rt);
    long long w0t = 64 * 128;
    pad_rows_kernel<<<cdiv(w0t, 256), 256, 0, stream>>>(P(76), w0p, w0t, 43, 128);
    gemm(stream, rawb, w0p, P(77), t128, N, 64, 128, 1, wtb);
    bn(stream, t128, N, 128, P(78), P(79), mean, rstd, part);
    tobf(stream, t128, t128b, (long long)N * 128);
    linear_bn(stream, t128b, P(80), P(81), P(82), P(83), x0, N, 128, 64, mean, rstd, part, wtb);
    tobf(stream, x0, x0b, (long long)N * 64);

    // ---- gcu1: cin=64, h=128, fuse 256->256 ----
    edge_conv(stream, x0b, N, 64, 128, tpl_src, tpl_dst, E_,
              P(28), P(29), P(30), P(31), P(32), P(33), P(34), P(35),
              convT, A1, B1, e1, e1b, e2, wdiff, mean, rstd, part, wtb);
    edge_conv(stream, x0b, N, 64, 128, geo_src, geo_dst, E_,
              P(20), P(21), P(22), P(23), P(24), P(25), P(26), P(27),
              convG, A1, B1, e1, e1b, e2, wdiff, mean, rstd, part, wtb);
    long long ct1 = (long long)N * 256;
    concat2_kernel<<<cdiv(ct1, 256), 256, 0, stream>>>(convT, convG, catH, ct1, 128, 128);
    tobf(stream, catH, catHb, ct1);
    linear_bn(stream, catHb, P(16), P(17), P(18), P(19), x1, N, 256, 256, mean, rstd, part, wtb);
    tobf(stream, x1, x1b, ct1);

    // ---- mlp2: 256 -> 256 -> 1024 ; global max-pool per batch ----
    linear_bn(stream, x1b, P(84), P(85), P(86), P(87), convT, N, 256, 256, mean, rstd, part, wtb);
    tobf(stream, convT, tmpb, ct1);
    linear_bn(stream, tmpb, P(88), P(89), P(90), P(91), g, N, 256, 1024, mean, rstd, part, wtb);
    long long gm = (long long)B_ * 1024;
    fill_kernel<<<cdiv(gm, 256), 256, 0, stream>>>(gmax, gm, -INFINITY);
    long long gt = (long long)N * 1024;
    seg_max_kernel<<<cdiv(gt, 256), 256, 0, stream>>>(g, batch, gmax, gt, 1024);
    fix_neginf_kernel<<<cdiv(gm, 256), 256, 0, stream>>>(gmax, gm);
    gather_rows_kernel<<<cdiv(gt, 256), 256, 0, stream>>>(gmax, batch, gb, gt, 1024);

    // ---- gcu2: cin=256, h=256, fuse 512->512 ----
    edge_conv(stream, x1b, N, 256, 256, tpl_src, tpl_dst, E_,
              P(48), P(49), P(50), P(51), P(52), P(53), P(54), P(55),
              convT, A1, B1, e1, e1b, e2, wdiff, mean, rstd, part, wtb);
    edge_conv(stream, x1b, N, 256, 256, geo_src, geo_dst, E_,
              P(40), P(41), P(42), P(43), P(44), P(45), P(46), P(47),
              convG, A1, B1, e1, e1b, e2, wdiff, mean, rstd, part, wtb);
    long long ct2 = (long long)N * 512;
    concat2_kernel<<<cdiv(ct2, 256), 256, 0, stream>>>(convT, convG, catH, ct2, 256, 256);
    tobf(stream, catH, catHb, ct2);
    linear_bn(stream, catHb, P(36), P(37), P(38), P(39), x2, N, 512, 512, mean, rstd, part, wtb);
    tobf(stream, x2, x2b, ct2);

    // ---- gcu3: cin=512, h=256, fuse 512->512 ----
    edge_conv(stream, x2b, N, 512, 256, tpl_src, tpl_dst, E_,
              P(68), P(69), P(70), P(71), P(72), P(73), P(74), P(75),
              convT, A1, B1, e1, e1b, e2, wdiff, mean, rstd, part, wtb);
    edge_conv(stream, x2b, N, 512, 256, geo_src, geo_dst, E_,
              P(60), P(61), P(62), P(63), P(64), P(65), P(66), P(67),
              convG, A1, B1, e1, e1b, e2, wdiff, mean, rstd, part, wtb);
    concat2_kernel<<<cdiv(ct2, 256), 256, 0, stream>>>(convT, convG, catH, ct2, 256, 256);
    tobf(stream, catH, catHb, ct2);
    linear_bn(stream, catHb, P(56), P(57), P(58), P(59), x3, N, 512, 512, mean, rstd, part, wtb);

    // ---- classifier: concat(x3, gb) -> 1024 -> 512 -> 256 -> 5 (pad 64) ----
    long long ft = (long long)N * 1536;
    concat2_kernel<<<cdiv(ft, 256), 256, 0, stream>>>(x3, gb, feat, ft, 512, 1024);
    tobf(stream, feat, featb, ft);
    linear_bn(stream, featb, P(2),  P(3),  P(4),  P(5),  c1, N, 1536, 1024, mean, rstd, part, wtb);
    tobf(stream, c1, c1b, (long long)N * 1024);
    linear_bn(stream, c1b, P(6),  P(7),  P(8),  P(9),  c2, N, 1024, 512, mean, rstd, part, wtb);
    tobf(stream, c2, c2b, (long long)N * 512);
    linear_bn(stream, c2b, P(10), P(11), P(12), P(13), c3, N, 512, 256, mean, rstd, part, wtb);
    tobf(stream, c3, c3b, (long long)N * 256);
    long long wft = 256 * 64;
    pad_cols_kernel<<<cdiv(wft, 256), 256, 0, stream>>>(P(14), wfp, wft, 5, 64);
    pad_cols_kernel<<<1, 64, 0, stream>>>(P(15), bfp, 64, 5, 64);
    gemm(stream, c3b, wfp, bfp, cout, N, 256, 64, 0, wtb);
    long long ot = (long long)N * 5;
    extract_cols_kernel<<<cdiv(ot, 256), 256, 0, stream>>>(cout, (float*)d_out, ot, 5, 64);
}
